// BiLSTMCRFModel_24584392802476
// MI455X (gfx1250) — matrix operand order repaired and checked by reading, not compiled, not run on hardware
//
#include <hip/hip_runtime.h>
#include <hip/hip_bf16.h>

typedef __attribute__((ext_vector_type(16))) _Float16 v16h;
typedef __attribute__((ext_vector_type(8)))  float    v8f;

#define B_SEQ 8192
#define T_LEN 256
#define K_TAGS 16

#define LOG2E 1.4426950408889634f
#define LN2   0.6931471805599453f

// Fast range-safe transcendentals: raw v_exp_f32 / v_log_f32.
// Safe here: exp args are <= 0 (post max-rescale), log args are >= ~0.5
// (dot always contains the max element's term), so no denorm/overflow paths.
__device__ __forceinline__ float fast_exp(float x) {
    return __builtin_amdgcn_exp2f(x * LOG2E);
}
__device__ __forceinline__ float fast_log(float x) {
    return __builtin_amdgcn_logf(x) * LN2;
}

// ---------------------------------------------------------------------------
// Forward (log_Z) kernel: one wave (32 lanes) handles 16 sequences.
// Per time step: alpha'[s][j] = em[s][t][j] + m_s + log( p_s . Etr_{tidx_s} )
// computed as one 16x64 (A) x 64x16 (B) f16 WMMA pair with f32 accumulation.
// B stacks exp() of the 4 transition matrices; A zero-masks each sequence's
// exp(alpha - max) into its tidx block, so no redundant matmuls are needed.
// ---------------------------------------------------------------------------
__global__ __launch_bounds__(32)
void crf_forward_wmma_kernel(const float* __restrict__ emissions,
                             const int*   __restrict__ who2who,
                             const int*   __restrict__ init_count,
                             const int*   __restrict__ distance,
                             const float* __restrict__ trans_stack,
                             const float* __restrict__ start_scores,
                             float*       __restrict__ logZ_out)
{
    __shared__ float alpha[16][17];   // [seq][tag], padded vs bank conflicts
    __shared__ float pbuf[16][17];    // exp(alpha - m)
    __shared__ float mbuf[16];        // per-seq max
    __shared__ float hmax[2][16];     // half-wave partial maxima
    __shared__ int   tbuf[16];        // per-seq transition index

    const int lane = threadIdx.x;     // 0..31
    const int hi   = lane >> 4;       // which 16-lane half
    const int lid  = lane & 15;
    const int b0   = blockIdx.x * 16; // first sequence of this group

    // ---- Constant B fragments: 64x16 f16, row 16m+i = exp(trans[m][i][:]).
    // B k order used here, same as A (wave32, K=32/WMMA, 8 VGPRs): lanes 0-15 hold K=0..15
    // (col N = lane), lanes 16-31 hold K=16..31; element e <-> K = Kbase + e.
    v16h bfrag0, bfrag1;
    {
        const int N = lid;
        const int kbase = hi * 16;
        #pragma unroll
        for (int e = 0; e < 16; ++e) {
            int kg0 = kbase + e;        // K row for WMMA #0 (K 0..31)
            int kg1 = 32 + kbase + e;   // K row for WMMA #1 (K 32..63)
            bfrag0[e] = (_Float16)fast_exp(trans_stack[(((kg0 >> 4) * 16) + (kg0 & 15)) * 16 + N]);
            bfrag1[e] = (_Float16)fast_exp(trans_stack[(((kg1 >> 4) * 16) + (kg1 & 15)) * 16 + N]);
        }
    }

    // ---- alpha0 = start_scores + emissions[:,0,:] ----
    #pragma unroll
    for (int jj = 0; jj < 8; ++jj) {
        int j = hi * 8 + jj;
        alpha[lid][j] = start_scores[j] +
                        emissions[((size_t)(b0 + lid) * T_LEN) * K_TAGS + j];
    }
    __syncthreads();

    for (int t = 1; t < T_LEN; ++t) {
        // Prefetch next step's emission tile (16 rows, 16 KB apart) into
        // cache one iteration ahead: emits global_prefetch_b8.
        {
            int tn = (t + 1 < T_LEN) ? (t + 1) : (T_LEN - 1);
            __builtin_prefetch(
                &emissions[((size_t)(b0 + lid) * T_LEN + tn) * K_TAGS], 0, 3);
        }

        // Phase A: per-sequence max + exp, split across the two lane halves.
        float av[8];
        float pm = -3.4e38f;
        #pragma unroll
        for (int jj = 0; jj < 8; ++jj) {
            av[jj] = alpha[lid][hi * 8 + jj];
            pm = fmaxf(pm, av[jj]);
        }
        hmax[hi][lid] = pm;
        if (hi == 0) {
            int idx = (b0 + lid) * (T_LEN - 1) + (t - 1);
            int w  = who2who[idx];
            int ic = init_count[idx];
            int dd = distance[idx];
            tbuf[lid] = (w == 1) ? 0 : (ic == 0 ? 1 : ((dd == 0) ? 2 : 3));
        }
        __syncthreads();
        float m = fmaxf(hmax[0][lid], hmax[1][lid]);
        if (hi == 0) mbuf[lid] = m;
        #pragma unroll
        for (int jj = 0; jj < 8; ++jj)
            pbuf[lid][hi * 8 + jj] = fast_exp(av[jj] - m);
        __syncthreads();

        // Phase B: build A fragments (16x64, zero outside the tidx block).
        // A k order used here, same as B (16x32/WMMA): lane holds row M = lane%16;
        // element e <-> K = e + 16*hi   (+32 for WMMA #1).
        const int tsel = tbuf[lid];
        v16h afrag0, afrag1;
        #pragma unroll
        for (int e = 0; e < 16; ++e) {
            int kg0 = e + 16 * hi;
            int kg1 = kg0 + 32;
            afrag0[e] = ((kg0 >> 4) == tsel) ? (_Float16)pbuf[lid][kg0 & 15]
                                             : (_Float16)0.0f;
            afrag1[e] = ((kg1 >> 4) == tsel) ? (_Float16)pbuf[lid][kg1 & 15]
                                             : (_Float16)0.0f;
        }

        // Phase C: D = A0*B0 + A1*B1  (chained v_wmma_f32_16x16x32_f16)
        v8f acc = {};
        acc = __builtin_amdgcn_wmma_f32_16x16x32_f16(
                  false, afrag0, false, bfrag0, (short)0, acc, false, false);
        acc = __builtin_amdgcn_wmma_f32_16x16x32_f16(
                  false, afrag1, false, bfrag1, (short)0, acc, false, false);
        __syncthreads();  // done reading pbuf/alpha before overwriting alpha

        // Phase D: alpha'[s][j] = em + m_s + log(D[s][j]).
        // D layout: element r <-> row M = r + 8*hi, col N = lane%16.
        #pragma unroll
        for (int r = 0; r < 8; ++r) {
            int s = r + 8 * hi;
            float e = emissions[((size_t)(b0 + s) * T_LEN + t) * K_TAGS + lid];
            alpha[s][lid] = e + mbuf[s] + fast_log(acc[r]);
        }
        __syncthreads();
    }

    // ---- log_Z[s] = logsumexp_j alpha[s][:] ----
    {
        float av[8];
        float pm = -3.4e38f;
        #pragma unroll
        for (int jj = 0; jj < 8; ++jj) {
            av[jj] = alpha[lid][hi * 8 + jj];
            pm = fmaxf(pm, av[jj]);
        }
        hmax[hi][lid] = pm;
        __syncthreads();
        float m = fmaxf(hmax[0][lid], hmax[1][lid]);
        float sum = 0.f;
        #pragma unroll
        for (int jj = 0; jj < 8; ++jj) sum += fast_exp(av[jj] - m);
        pbuf[hi][lid] = sum;  // reuse as [2][16] partial sums
        __syncthreads();
        if (hi == 0)
            logZ_out[b0 + lid] = m + fast_log(pbuf[0][lid] + pbuf[1][lid]);
    }
}

// ---------------------------------------------------------------------------
// Gold path score: one thread per sequence (pure gather + FMA stream).
// ---------------------------------------------------------------------------
__global__ __launch_bounds__(256)
void crf_gold_kernel(const float* __restrict__ emissions,
                     const int*   __restrict__ tags,
                     const int*   __restrict__ who2who,
                     const int*   __restrict__ init_count,
                     const int*   __restrict__ distance,
                     const float* __restrict__ class_weights,
                     const float* __restrict__ trans_stack,
                     const float* __restrict__ start_scores,
                     float*       __restrict__ gold_out)
{
    int b = blockIdx.x * blockDim.x + threadIdx.x;
    if (b >= B_SEQ) return;
    const int* tg = tags + (size_t)b * T_LEN;
    int prev = tg[0];
    float g = class_weights[prev] *
              (start_scores[prev] + emissions[((size_t)b * T_LEN) * K_TAGS + prev]);
    for (int t = 1; t < T_LEN; ++t) {
        int cur = tg[t];
        int idx = b * (T_LEN - 1) + (t - 1);
        int w  = who2who[idx];
        int ic = init_count[idx];
        int dd = distance[idx];
        int ti = (w == 1) ? 0 : (ic == 0 ? 1 : ((dd == 0) ? 2 : 3));
        float tr = trans_stack[(ti * K_TAGS + prev) * K_TAGS + cur];
        float e  = emissions[((size_t)b * T_LEN + t) * K_TAGS + cur];
        g = fmaf(class_weights[cur], tr + e, g);
        prev = cur;
    }
    gold_out[b] = g;
}

extern "C" void kernel_launch(void* const* d_in, const int* in_sizes, int n_in,
                              void* d_out, int out_size, void* d_ws, size_t ws_size,
                              hipStream_t stream) {
    const float* emissions     = (const float*)d_in[0];
    const int*   tags          = (const int*)  d_in[1];
    const int*   who2who       = (const int*)  d_in[2];
    const int*   init_count    = (const int*)  d_in[3];
    const int*   distance      = (const int*)  d_in[4];
    const float* class_weights = (const float*)d_in[5];
    const float* trans_stack   = (const float*)d_in[6];
    const float* start_scores  = (const float*)d_in[7];

    float* gold = (float*)d_out;          // outputs concatenated: gold, log_Z
    float* logZ = (float*)d_out + B_SEQ;

    crf_gold_kernel<<<B_SEQ / 256, 256, 0, stream>>>(
        emissions, tags, who2who, init_count, distance,
        class_weights, trans_stack, start_scores, gold);

    crf_forward_wmma_kernel<<<B_SEQ / 16, 32, 0, stream>>>(
        emissions, who2who, init_count, distance,
        trans_stack, start_scores, logZ);
}
